// CRF_3573412790840
// MI455X (gfx1250) — compile-verified, matrix operand order checked by reading, not run on hardware
//
#include <hip/hip_runtime.h>
#include <hip/hip_bf16.h>
#include <math.h>

// CRF forward pass on MI455X (gfx1250, wave32, WMMA).
//
// logsumexp recurrence in normalized-exp domain:
//   S = u x E  (WMMA, E = exp(T) loop-invariant f16 in registers)
//   q = S * exp(em);  M += log(max q)  [1 log/row/step];  u = q / max q  (f16)
// Memory-bound: 64MB emissions @ 23.3 TB/s ~= 2.7us floor. Emissions are
// double-buffered in LDS via global_load_async_to_lds_b128 issued one step
// ahead (ASYNCcnt-tracked, drained with s_wait_asynccnt), so the per-step
// critical path is WMMA + a handful of VALU ops.

typedef __attribute__((ext_vector_type(16))) _Float16 v16h;
typedef __attribute__((ext_vector_type(8)))  float    v8f;

#define SEQ   512
#define BATCH 512
#define TAGS  64
#define ROWS  16      // batch rows per block (WMMA M)
#define NTHREADS 128  // 4 wave32s; each wave owns one 16-col N-tile

union AFrag { v16h v; float4 f4[2]; };

// Load a 16-bit WMMA A/B fragment (16x32 K-chunk) from a row-major [*,64] f16 LDS
// buffer, matching the CDNA5 A-matrix layout:
//   lanes 0-15 : v0..3 = K 0..7,  v4..7 = K 16..23   (relative to kc*32)
//   lanes 16-31: v0..3 = K 8..15, v4..7 = K 24..31
__device__ __forceinline__ v16h load_frag(const _Float16* base, int row, int kc, int half) {
    AFrag a;
    const _Float16* p = base + row * 64 + kc * 32 + half * 8;
    a.f4[0] = *(const float4*)(p);        // ds_load_b128
    a.f4[1] = *(const float4*)(p + 16);   // ds_load_b128
    return a.v;
}

// Per-lane async DMA: LDS[lds_off .. +15] = MEM[gaddr .. +15]  (ASYNCcnt++)
__device__ __forceinline__ void async_load_b128(unsigned lds_off, const float* gaddr) {
    asm volatile("global_load_async_to_lds_b128 %0, %1, off"
                 :: "v"(lds_off), "v"((unsigned long long)(size_t)gaddr)
                 : "memory");
}
__device__ __forceinline__ unsigned lds_off_of(const void* p) {
    return (unsigned)(size_t)p;   // flat shared ptr: low 32 bits = LDS byte offset
}

__global__ __launch_bounds__(NTHREADS)
void crf_forward_kernel(const float* __restrict__ em,
                        const float* __restrict__ startT,
                        const float* __restrict__ endT,
                        const float* __restrict__ trans,
                        float* __restrict__ ws_den) {
    __shared__ _Float16 Et[TAGS * TAGS];       // Et[n][k] = exp(trans[k][n]) (f16 B operand)
    __shared__ _Float16 P16[ROWS * TAGS];      // u (normalized exp state, f16 A operand)
    __shared__ float    lpbuf[ROWS * TAGS];    // S (C layout) then q (row major)
    __shared__ float    embuf[2][ROWS * TAGS]; // async double buffer for emissions
    __shared__ float    pmax[8][ROWS];         // per-row partial maxima of q

    const int tid   = threadIdx.x;
    const int lane  = tid & 31;
    const int wave  = tid >> 5;                // N-tile id 0..3
    const int bbase = blockIdx.x * ROWS;

    // row-major work assignment: 8 threads per row, 8 contiguous cols each
    const int row = tid & 15;
    const int cg  = tid >> 4;   // 0..7
    const int cb  = cg * 8;

    // kick off async DMA of em[1] into embuf[1] as early as possible
    {
        const float* g = em + (size_t)1 * BATCH * TAGS + (size_t)(bbase + row) * TAGS + cb;
        unsigned lo = lds_off_of(&embuf[1][row * 64 + cb]);
        async_load_b128(lo,      g);
        async_load_b128(lo + 16, g + 4);
    }

    // ---- precompute Et = exp(T)^T in f16 (loop-invariant B matrix) ----
    for (int e = tid; e < TAGS * TAGS; e += NTHREADS) {
        int k = e >> 6, n = e & 63;
        Et[n * 64 + k] = (_Float16)__expf(trans[e]);
    }
    __syncthreads();

    // ---- preload this wave's B fragments into registers (held across 511 steps) ----
    const int arow  = lane & 15;
    const int ahalf = lane >> 4;
    v16h b0 = load_frag(Et, wave * 16 + arow, 0, ahalf);
    v16h b1 = load_frag(Et, wave * 16 + arow, 1, ahalf);

    float qreg[8];
    float Mrow = 0.f;
    {   // init: q0 = exp(start + em[0]);  lp0 = log q0
        const float4* e4 = (const float4*)(em + (size_t)(bbase + row) * TAGS + cb);
        float4 ea = e4[0], eb = e4[1];
        const float4* s4 = (const float4*)(startT + cb);
        float4 sa = s4[0], sb = s4[1];
        float v[8] = { ea.x + sa.x, ea.y + sa.y, ea.z + sa.z, ea.w + sa.w,
                       eb.x + sb.x, eb.y + sb.y, eb.z + sb.z, eb.w + sb.w };
        float mloc = -INFINITY;
#pragma unroll
        for (int k = 0; k < 8; ++k) {
            float q = __expf(v[k]);
            qreg[k] = q;
            lpbuf[row * 64 + cb + k] = q;
            mloc = fmaxf(mloc, q);
        }
        pmax[cg][row] = mloc;
    }

    for (int t = 1; t < SEQ; ++t) {
        __syncthreads();
        // ---- stage B: maxq; M += log(maxq); u = q * rcp(maxq) -> f16 LDS ----
        float maxq = pmax[0][row];
#pragma unroll
        for (int g = 1; g < 8; ++g) maxq = fmaxf(maxq, pmax[g][row]);
        Mrow += __logf(maxq);                       // the ONLY log in the loop
        float inv = __builtin_amdgcn_rcpf(maxq);
#pragma unroll
        for (int k = 0; k < 8; ++k)
            P16[row * 64 + cb + k] = (_Float16)(qreg[k] * inv);
        __syncthreads();

        // ---- WMMA stage: S(16x16 tile) = u(16x64) x E(64x16), K split in two ----
        v16h a0 = load_frag(P16, arow, 0, ahalf);
        v16h a1 = load_frag(P16, arow, 1, ahalf);
        v8f c = {};
        c = __builtin_amdgcn_wmma_f32_16x16x32_f16(false, a0, false, b0, (short)0, c, false, false);
        c = __builtin_amdgcn_wmma_f32_16x16x32_f16(false, a1, false, b1, (short)0, c, false, false);

        // raw S back to LDS in the f32 C layout (VGPR r -> row r + 8*half, col = lane&15)
        const int ccol = wave * 16 + (lane & 15);
#pragma unroll
        for (int r = 0; r < 8; ++r)
            lpbuf[(r + 8 * ahalf) * 64 + ccol] = c[r];

        // async DMA next timestep's emissions (clamped reload on last iter; unread)
        {
            int tn = (t + 1 < SEQ) ? (t + 1) : (SEQ - 1);
            const float* g = em + (size_t)tn * BATCH * TAGS + (size_t)(bbase + row) * TAGS + cb;
            unsigned lo = lds_off_of(&embuf[(t + 1) & 1][row * 64 + cb]);
            async_load_b128(lo,      g);
            async_load_b128(lo + 16, g + 4);
        }
        if (t + 2 < SEQ)   // keep L2 warm two steps ahead
            __builtin_prefetch(em + (size_t)(t + 2) * BATCH * TAGS
                                  + (size_t)(bbase + row) * TAGS + cb, 0, 1);
        __syncthreads();

        // ---- stage A: q_t = S * exp(em_t)  (emissions from async LDS buffer) ----
        asm volatile("s_wait_asynccnt 0x2" ::: "memory");  // drain this step's DMA
        const float4* e4 = (const float4*)&embuf[t & 1][row * 64 + cb];
        float4 ea = e4[0], eb = e4[1];
        float ev[8] = { ea.x, ea.y, ea.z, ea.w, eb.x, eb.y, eb.z, eb.w };
        float mloc = -INFINITY;
#pragma unroll
        for (int k = 0; k < 8; ++k) {
            float q = lpbuf[row * 64 + cb + k] * __expf(ev[k]);
            qreg[k] = q;
            lpbuf[row * 64 + cb + k] = q;
            mloc = fmaxf(mloc, q);
        }
        pmax[cg][row] = mloc;
    }
    __syncthreads();

    // ---- den[b] = M_510 + log( sum_j q_511[j] * exp(end[j]) )  (fixed order) ----
    if (tid < ROWS) {
        float s = 0.f;
        for (int j = 0; j < TAGS; ++j)
            s += lpbuf[tid * 64 + j] * __expf(endT[j]);
        ws_den[bbase + tid] = Mrow + __logf(s);
    }
}

// Numerator: gather path score per batch element (mask is all ones in this problem).
__global__ void crf_joint_kernel(const float* __restrict__ em,
                                 const float* __restrict__ startT,
                                 const float* __restrict__ endT,
                                 const float* __restrict__ trans,
                                 const int* __restrict__ tags,
                                 float* __restrict__ ws_num) {
    int b = blockIdx.x * blockDim.x + threadIdx.x;
    if (b >= BATCH) return;
    int tg = tags[b];
    float acc = startT[tg];
    for (int t = 0; t < SEQ - 1; ++t) {
        int tg1 = tags[(t + 1) * BATCH + b];
        acc += em[((size_t)t * BATCH + b) * TAGS + tg];
        acc += trans[tg * TAGS + tg1];
        tg = tg1;
    }
    acc += endT[tg] + em[((size_t)(SEQ - 1) * BATCH + b) * TAGS + tg];
    ws_num[b] = acc;
}

// Deterministic fixed-order reduction: out = sum_b (num[b] - den[b]).
__global__ void crf_reduce_kernel(const float* __restrict__ ws_num,
                                  const float* __restrict__ ws_den,
                                  float* __restrict__ out) {
    __shared__ float sm[256];
    float a = 0.f;
    for (int b = threadIdx.x; b < BATCH; b += 256) a += ws_num[b] - ws_den[b];
    sm[threadIdx.x] = a;
    __syncthreads();
    for (int s = 128; s > 0; s >>= 1) {
        if (threadIdx.x < s) sm[threadIdx.x] += sm[threadIdx.x + s];
        __syncthreads();
    }
    if (threadIdx.x == 0) out[0] = sm[0];
}

extern "C" void kernel_launch(void* const* d_in, const int* in_sizes, int n_in,
                              void* d_out, int out_size, void* d_ws, size_t ws_size,
                              hipStream_t stream) {
    const float* em   = (const float*)d_in[0];  // [SEQ, BATCH, TAGS]
    const float* st   = (const float*)d_in[1];  // [TAGS]
    const float* en   = (const float*)d_in[2];  // [TAGS]
    const float* tr   = (const float*)d_in[3];  // [TAGS, TAGS]
    const int*   tags = (const int*)d_in[4];    // [SEQ, BATCH]
    // d_in[5] = mask (all ones) -- unused

    float* ws_den = (float*)d_ws;         // [BATCH]
    float* ws_num = ws_den + BATCH;       // [BATCH]

    crf_forward_kernel<<<BATCH / ROWS, NTHREADS, 0, stream>>>(em, st, en, tr, ws_den);
    crf_joint_kernel<<<(BATCH + 255) / 256, 256, 0, stream>>>(em, st, en, tr, tags, ws_num);
    crf_reduce_kernel<<<1, 256, 0, stream>>>(ws_num, ws_den, (float*)d_out);
}
